// PositionAttention_86165633892446
// MI455X (gfx1250) — matrix operand order repaired and checked by reading, not compiled, not run on hardware
//
#include <hip/hip_runtime.h>
#include <cmath>

// ---------------------------------------------------------------------------
// PositionAttention fused kernel for MI455X (gfx1250, wave32, WMMA)
//   B=4, C=128, N=8192, CQK=32.  QK dim padded 32(+3 pos)->64 for bf16 WMMA.
// Stage 1: proj_qk  -> Qp,Kp  bf16 [B][N][64]   (pos-bias folded as channels 32..34)
// Stage 2: proj_v   -> Vbf    bf16 [B][C][N]
// Stage 3: attn     -> flash-attention per 16-query tile per wave,
//                      4 waves/WG share double-buffered K/V tiles in LDS,
//                      filled by GLOBAL_LOAD_ASYNC_TO_LDS_B128 (ASYNCcnt DMA).
//   Row-sums via a 9th all-ones WMMA tile; row-max via v_permlane16_b32.
// ---------------------------------------------------------------------------

#define PA_USE_ASYNC_LDS 1

typedef unsigned short u16;
typedef __attribute__((ext_vector_type(16))) __bf16 v16bf;
typedef __attribute__((ext_vector_type(8)))  float  v8f;

static constexpr int BB  = 4;
static constexpr int CC  = 128;
static constexpr int NN  = 8192;
static constexpr int KD  = 64;     // padded q/k channel dim

union Op16 { uint4 q[2]; v16bf v; };

__device__ __forceinline__ u16 f2bf(float f) {
    union { float f; unsigned u; } v; v.f = f;
    unsigned r = v.u + 0x7FFFu + ((v.u >> 16) & 1u);   // round-to-nearest-even
    return (u16)(r >> 16);
}

// Async DMA: copy 16 bytes per lane, global -> LDS, tracked by ASYNCcnt.
__device__ __forceinline__ void async_g2l_16B(unsigned lds_off, const void* gaddr) {
#if PA_USE_ASYNC_LDS
    asm volatile("global_load_async_to_lds_b128 %0, %1, off"
                 :: "v"(lds_off), "v"(gaddr) : "memory");
#endif
}
__device__ __forceinline__ void async_wait0() {
#if PA_USE_ASYNC_LDS
    asm volatile("s_wait_asynccnt 0x0" ::: "memory");
#endif
}
// generic LDS pointer -> 32-bit LDS byte offset (AS3 offset = low 32 bits)
__device__ __forceinline__ unsigned lds_off32(const void* p) {
    return (unsigned)(unsigned long long)p;
}

// max-reduce across each 16-lane half of the wave (rows of the WMMA C layout)
__device__ __forceinline__ float rowmax16(float v) {
#if defined(__has_builtin) && __has_builtin(__builtin_amdgcn_permlane16)
    int x;
    x = __builtin_amdgcn_permlane16(__float_as_int(v), __float_as_int(v),
                                    0x67452301u, 0xEFCDAB89u, false, false);   // xor 1
    v = fmaxf(v, __int_as_float(x));
    x = __builtin_amdgcn_permlane16(__float_as_int(v), __float_as_int(v),
                                    0x54761032u, 0xDCFE98BAu, false, false);   // xor 2
    v = fmaxf(v, __int_as_float(x));
    x = __builtin_amdgcn_permlane16(__float_as_int(v), __float_as_int(v),
                                    0x32107654u, 0xBA98FEDCu, false, false);   // xor 4
    v = fmaxf(v, __int_as_float(x));
    x = __builtin_amdgcn_permlane16(__float_as_int(v), __float_as_int(v),
                                    0xFEDCBA98u, 0x76543210u, false, false);   // xor 8
    v = fmaxf(v, __int_as_float(x));
#else
    v = fmaxf(v, __shfl_xor(v, 1, 32));
    v = fmaxf(v, __shfl_xor(v, 2, 32));
    v = fmaxf(v, __shfl_xor(v, 4, 32));
    v = fmaxf(v, __shfl_xor(v, 8, 32));
#endif
    return v;
}

// --------------------------- Stage 1: Q/K projection ------------------------
__global__ void pa_proj_qk(const float* __restrict__ x, const float* __restrict__ xyz,
                           const float* __restrict__ Wq, const float* __restrict__ bq,
                           const float* __restrict__ Wk, const float* __restrict__ bk,
                           u16* __restrict__ Qp, u16* __restrict__ Kp) {
    __shared__ float sWq[32 * 128];
    __shared__ float sWk[32 * 128];
    const int tid = threadIdx.x;
    for (int i = tid; i < 32 * 128; i += 128) { sWq[i] = Wq[i]; sWk[i] = Wk[i]; }
    __syncthreads();

    const int idx = blockIdx.x * 128 + tid;     // flat (b, n)
    const int b = idx >> 13;                    // / 8192
    const int n = idx & (NN - 1);

    float aq[32], ak[32];
#pragma unroll
    for (int o = 0; o < 32; ++o) { aq[o] = bq[o]; ak[o] = bk[o]; }

    for (int c = 0; c < CC; ++c) {
        const float xv = x[((size_t)b * CC + c) * NN + n];
#pragma unroll
        for (int o = 0; o < 32; ++o) {
            aq[o] = fmaf(sWq[o * 128 + c], xv, aq[o]);
            ak[o] = fmaf(sWk[o * 128 + c], xv, ak[o]);
        }
    }

    const float px = xyz[((size_t)b * NN + n) * 3 + 0];
    const float py = xyz[((size_t)b * NN + n) * 3 + 1];
    const float pz = xyz[((size_t)b * NN + n) * 3 + 2];
    const float invn = 1.0f / (sqrtf(px * px + py * py + pz * pz) + 1e-8f);

    u16* qd = Qp + (size_t)idx * KD;
    u16* kd = Kp + (size_t)idx * KD;
#pragma unroll
    for (int o = 0; o < 32; ++o) { qd[o] = f2bf(aq[o]); kd[o] = f2bf(ak[o]); }
    qd[32] = f2bf(0.1f * px * invn); kd[32] = f2bf(px * invn);
    qd[33] = f2bf(0.1f * py * invn); kd[33] = f2bf(py * invn);
    qd[34] = f2bf(0.1f * pz * invn); kd[34] = f2bf(pz * invn);
#pragma unroll
    for (int o = 35; o < KD; ++o) { qd[o] = 0; kd[o] = 0; }
}

// --------------------------- Stage 2: V projection --------------------------
__global__ void pa_proj_v(const float* __restrict__ x, const float* __restrict__ Wv,
                          const float* __restrict__ bv, u16* __restrict__ Vbf) {
    __shared__ float sW[128 * 128];             // 64 KB
    const int tid = threadIdx.x;
    for (int i = tid; i < 128 * 128; i += 128) sW[i] = Wv[i];
    __syncthreads();

    const int idx = blockIdx.x * 128 + tid;
    const int b = idx >> 13;
    const int n = idx & (NN - 1);

    for (int cb = 0; cb < CC; cb += 16) {
        float acc[16];
#pragma unroll
        for (int j = 0; j < 16; ++j) acc[j] = bv[cb + j];
        for (int ci = 0; ci < CC; ++ci) {
            const float xv = x[((size_t)b * CC + ci) * NN + n];
#pragma unroll
            for (int j = 0; j < 16; ++j)
                acc[j] = fmaf(sW[(cb + j) * 128 + ci], xv, acc[j]);
        }
#pragma unroll
        for (int j = 0; j < 16; ++j)
            Vbf[((size_t)b * CC + cb + j) * NN + n] = f2bf(acc[j]);
    }
}

// --------------------------- Stage 3: fused attention -----------------------
__global__ void __launch_bounds__(128)
pa_attn(const u16* __restrict__ Qp, const u16* __restrict__ Kp,
        const u16* __restrict__ Vbf, const float* __restrict__ x,
        const float* __restrict__ gamma, float* __restrict__ out) {
    __shared__ u16 Ks[2][32 * KD];     // 2 x 4 KB : [key][ch]
    __shared__ u16 Vs[2][CC * 32];     // 2 x 8 KB : [ch][key]
    __shared__ u16 Ps[4][16 * 32];     // 4 KB     : per-wave [qrow][key]

    const int tid  = threadIdx.x;
    const int wave = tid >> 5;
    const int lane = tid & 31;
    const int col  = lane & 15;        // N-col / A-row index
    const int hi   = lane >> 4;

    const int b     = blockIdx.x / (NN / 64);
    const int qt    = blockIdx.x % (NN / 64);
    const int qbase = qt * 64 + wave * 16;

    // per-thread DMA slots (constant across iterations)
    const unsigned lk0 = lds_off32(&Ks[0][tid * 8]);           // 16 B chunk
    const unsigned lv0 = lds_off32(&Vs[0][tid * 32]);          // 64 B row
    const unsigned kbufstride = lds_off32(&Ks[1][0]) - lds_off32(&Ks[0][0]);
    const unsigned vbufstride = lds_off32(&Vs[1][0]) - lds_off32(&Vs[0][0]);

    // issue async DMA of one 32-key K/V tile into buffer `buf`
    auto issue_tile = [&](int m, int buf) {
        const u16* kg = Kp + ((size_t)b * NN + m) * KD + tid * 8;
        const unsigned lk = lk0 + buf * kbufstride;
        async_g2l_16B(lk, kg);                                 // chunk 0 (2 KB)
        async_g2l_16B(lk + 2048, kg + 1024);                   // chunk 1 (2 KB)
        const u16* vg = Vbf + ((size_t)b * CC + tid) * NN + m;
        const unsigned lv = lv0 + buf * vbufstride;
        async_g2l_16B(lv,      vg);
        async_g2l_16B(lv + 16, vg + 8);
        async_g2l_16B(lv + 32, vg + 16);
        async_g2l_16B(lv + 48, vg + 24);
    };

    // ---- Q A-operands (held in VGPRs for the whole kernel) ----
    const u16* qrow = Qp + ((size_t)b * NN + qbase + col) * KD;
    Op16 qa0, qa1;
    qa0.q[0] = *(const uint4*)(qrow + hi * 8);
    qa0.q[1] = *(const uint4*)(qrow + hi * 8 + 16);
    qa1.q[0] = *(const uint4*)(qrow + 32 + hi * 8);
    qa1.q[1] = *(const uint4*)(qrow + 32 + hi * 8 + 16);

    // all-ones bf16 B operand -> 9th tile accumulates row sums of P
    Op16 ones;
    ones.q[0] = make_uint4(0x3F803F80u, 0x3F803F80u, 0x3F803F80u, 0x3F803F80u);
    ones.q[1] = ones.q[0];

    v8f o[9];
#pragma unroll
    for (int t = 0; t < 9; ++t) o[t] = (v8f){0, 0, 0, 0, 0, 0, 0, 0};
    float rm[8];
#pragma unroll
    for (int r = 0; r < 8; ++r) rm[r] = -1e30f;

    u16* Pw = Ps[wave];

#if PA_USE_ASYNC_LDS
    issue_tile(0, 0);
    async_wait0();
    __syncthreads();                               // buffer 0 ready for all waves
#endif

    for (int m = 0; m < NN; m += 32) {
        const int p = (m >> 5) & 1;

#if PA_USE_ASYNC_LDS
        if (m + 32 < NN) issue_tile(m + 32, p ^ 1);  // DMA next tile during compute
        const u16* Kc = Ks[p];
        const u16* Vc = Vs[p];
#else
        const uint4* ksrc = (const uint4*)(Kp + ((size_t)b * NN + m) * KD);
        const uint4 kr0 = ksrc[tid];
        const uint4 kr1 = ksrc[tid + 128];
        const uint4* vsrc = (const uint4*)(Vbf + ((size_t)b * CC + tid) * NN + m);
        const uint4 vr0 = vsrc[0], vr1 = vsrc[1], vr2 = vsrc[2], vr3 = vsrc[3];
        __syncthreads();
        ((uint4*)Ks[0])[tid]       = kr0;
        ((uint4*)Ks[0])[tid + 128] = kr1;
        uint4* vd = (uint4*)(Vs[0] + tid * 32);
        vd[0] = vr0; vd[1] = vr1; vd[2] = vr2; vd[3] = vr3;
        __syncthreads();
        const u16* Kc = Ks[0];
        const u16* Vc = Vs[0];
#endif

        // ---- S = Q'K' tiles (2 x 16q x 16k, K-dim 64) ----
        v8f s0 = (v8f){0, 0, 0, 0, 0, 0, 0, 0};
        v8f s1 = (v8f){0, 0, 0, 0, 0, 0, 0, 0};
        {
            const int k0 = col * KD + hi * 8;             // key tile 0, chunk 0
            Op16 kb; kb.q[0] = *(const uint4*)&Kc[k0];     kb.q[1] = *(const uint4*)&Kc[k0 + 16];
            s0 = __builtin_amdgcn_wmma_f32_16x16x32_bf16(false, qa0.v, false, kb.v, (short)0, s0, false, false);
            kb.q[0] = *(const uint4*)&Kc[k0 + 32];         kb.q[1] = *(const uint4*)&Kc[k0 + 48];
            s0 = __builtin_amdgcn_wmma_f32_16x16x32_bf16(false, qa1.v, false, kb.v, (short)0, s0, false, false);
            const int k1 = (16 + col) * KD + hi * 8;      // key tile 1
            kb.q[0] = *(const uint4*)&Kc[k1];              kb.q[1] = *(const uint4*)&Kc[k1 + 16];
            s1 = __builtin_amdgcn_wmma_f32_16x16x32_bf16(false, qa0.v, false, kb.v, (short)0, s1, false, false);
            kb.q[0] = *(const uint4*)&Kc[k1 + 32];         kb.q[1] = *(const uint4*)&Kc[k1 + 48];
            s1 = __builtin_amdgcn_wmma_f32_16x16x32_bf16(false, qa1.v, false, kb.v, (short)0, s1, false, false);
        }

        // ---- online softmax over the 32 new keys (rows live on half-waves) ----
        float scale[8];
#pragma unroll
        for (int r = 0; r < 8; ++r) {
            const float mx = rowmax16(fmaxf(s0[r], s1[r]));
            const float mn = fmaxf(rm[r], mx);
            const float p0 = __expf(s0[r] - mn);
            const float p1 = __expf(s1[r] - mn);
            scale[r] = __expf(rm[r] - mn);
            rm[r] = mn;
            const int row = r + 8 * hi;                    // C-layout row
            Pw[row * 32 + col]      = f2bf(p0);
            Pw[row * 32 + col + 16] = f2bf(p1);
        }
#pragma unroll
        for (int t = 0; t < 9; ++t)
#pragma unroll
            for (int r = 0; r < 8; ++r) o[t][r] *= scale[r];

        // ---- O += P * V  (A = P 16x32 from LDS, B = V 32x16 per channel tile)
        //      o[8] += P * ones  -> per-row sum of P (flash-attention denominator)
        Op16 pa;
        pa.q[0] = *(const uint4*)&Pw[col * 32 + hi * 8];
        pa.q[1] = *(const uint4*)&Pw[col * 32 + hi * 8 + 16];
#pragma unroll
        for (int t = 0; t < 8; ++t) {
            const int c = t * 16 + col;
            Op16 vb;
            vb.q[0] = *(const uint4*)&Vc[c * 32 + hi * 8];
            vb.q[1] = *(const uint4*)&Vc[c * 32 + hi * 8 + 16];
            o[t] = __builtin_amdgcn_wmma_f32_16x16x32_bf16(false, pa.v, false, vb.v, (short)0, o[t], false, false);
        }
        o[8] = __builtin_amdgcn_wmma_f32_16x16x32_bf16(false, pa.v, false, ones.v, (short)0, o[8], false, false);

#if PA_USE_ASYNC_LDS
        async_wait0();                              // next tile's DMA done
        __syncthreads();                            // all waves done with this tile
#endif
    }

    // ---- finalize: out = gamma * (O / rowsum) + x ----
    const float g = gamma[0];
    float inv[8];
#pragma unroll
    for (int r = 0; r < 8; ++r) inv[r] = 1.0f / o[8][r];   // every lane holds its rows' sum
#pragma unroll
    for (int t = 0; t < 8; ++t) {
        const int c = t * 16 + col;
        const size_t base = ((size_t)b * CC + c) * NN + qbase + 8 * hi;
#pragma unroll
        for (int r = 0; r < 8; ++r)
            out[base + r] = g * (o[t][r] * inv[r]) + x[base + r];
    }
}

// ---------------------------------------------------------------------------
extern "C" void kernel_launch(void* const* d_in, const int* in_sizes, int n_in,
                              void* d_out, int out_size, void* d_ws, size_t ws_size,
                              hipStream_t stream) {
    const float* x     = (const float*)d_in[0];
    const float* xyz   = (const float*)d_in[1];
    const float* Wq    = (const float*)d_in[2];
    const float* bq    = (const float*)d_in[3];
    const float* Wk    = (const float*)d_in[4];
    const float* bk    = (const float*)d_in[5];
    const float* Wv    = (const float*)d_in[6];
    const float* bv    = (const float*)d_in[7];
    const float* gamma = (const float*)d_in[8];
    float* out = (float*)d_out;

    u16* Qp  = (u16*)d_ws;                               // 4 MB
    u16* Kp  = Qp + (size_t)BB * NN * KD;                // 4 MB
    u16* Vbf = Kp + (size_t)BB * NN * KD;                // 8 MB

    const int ptBlocks = (BB * NN) / 128;                // 256
    pa_proj_qk<<<ptBlocks, 128, 0, stream>>>(x, xyz, Wq, bq, Wk, bk, Qp, Kp);
    pa_proj_v <<<ptBlocks, 128, 0, stream>>>(x, Wv, bv, Vbf);

    const int attnBlocks = BB * (NN / 64);               // 512
    pa_attn<<<attnBlocks, 128, 0, stream>>>(Qp, Kp, Vbf, x, gamma, out);
}
